// MyVectorQuantize_61950608278146
// MI455X (gfx1250) — matrix operand order repaired and checked by reading, not compiled, not run on hardware
//
#include <hip/hip_runtime.h>

// ---------------------------------------------------------------------------
// VQ forward for MI455X (gfx1250).
//   argmin_k ||z_m - e_k||^2  ==  argmin_k ( e_sq[k] - 2 * z_m . e_k )
// The M x K x D dot-product term runs on v_wmma_f32_16x16x32_bf16 (fp32
// accumulate). Codebook is pre-converted once to bf16 in workspace so the hot
// loop streams raw bf16 tiles via GLOBAL_LOAD_ASYNC_TO_LDS_B128 (double
// buffered, ASYNCcnt-synchronized). A-fragments are hoisted into registers
// across the whole K sweep; B-fragments are software-pipelined one K-step
// ahead so WMMAs never drain DScnt to zero. Gather / commit-loss: exact fp32.
// ---------------------------------------------------------------------------

typedef __attribute__((ext_vector_type(16))) __bf16        v16bf;
typedef __attribute__((ext_vector_type(8)))  float         v8f;
typedef __attribute__((ext_vector_type(4)))  float         f32x4;
typedef __attribute__((ext_vector_type(4)))  int           v4i;
typedef __attribute__((ext_vector_type(4)))  unsigned int  u32x4;
typedef __attribute__((ext_vector_type(2)))  unsigned int  u32x2;

// async-to-LDS builtin wants typed int4 pointers in explicit address spaces
typedef __attribute__((address_space(1))) v4i v4i_as1;   // global
typedef __attribute__((address_space(3))) v4i v4i_as3;   // LDS

#define D_DIM      512
#define BM         64
#define BN         64
#define LDS_STRIDE 520   // 512 bf16 + 8 pad -> 1040B row stride, conflict-free b128

#if defined(__has_builtin)
#if __has_builtin(__builtin_amdgcn_global_load_async_to_lds_b128)
#define HAVE_ASYNC_B128 1
#endif
#if __has_builtin(__builtin_amdgcn_s_wait_asynccnt)
#define HAVE_WAIT_ASYNC 1
#endif
#endif

__device__ __forceinline__ void wait_async_all() {
#if defined(HAVE_WAIT_ASYNC)
    __builtin_amdgcn_s_wait_asynccnt(0);
#elif defined(HAVE_ASYNC_B128)
    asm volatile("s_wait_asynccnt 0" ::: "memory");
#endif
}

__device__ __forceinline__ unsigned int f2bf(float f) {
    // round-to-nearest-even fp32 -> bf16
    unsigned int u = __float_as_uint(f);
    u += 0x7fffu + ((u >> 16) & 1u);
    return u >> 16;
}

__device__ __forceinline__ unsigned int orderF(float f) {
    // monotonic float -> uint mapping (total order for finite values)
    unsigned int u = __float_as_uint(f);
    return (u & 0x80000000u) ? ~u : (u | 0x80000000u);
}

// ---- codebook fp32 -> bf16 (once) fused with e_sq[k] = ||e_k||^2 ----------
extern "C" __global__ __launch_bounds__(128)
void vq_cvt_cb_kernel(const float* __restrict__ cb,
                      unsigned short* __restrict__ cb_bf,
                      float* __restrict__ esq) {
    const int k = blockIdx.x;
    const int t = threadIdx.x;
    f32x4 v = *(const f32x4*)(cb + (size_t)k * D_DIM + t * 4);
    u32x2 p;
    p.x = f2bf(v.x) | (f2bf(v.y) << 16);
    p.y = f2bf(v.z) | (f2bf(v.w) << 16);
    *(u32x2*)(cb_bf + (size_t)k * D_DIM + t * 4) = p;

    float s = v.x * v.x + v.y * v.y + v.z * v.z + v.w * v.w;
    #pragma unroll
    for (int off = 16; off > 0; off >>= 1)
        s += __shfl_down(s, off, 32);
    __shared__ float wsum[4];
    if ((t & 31) == 0) wsum[t >> 5] = s;
    __syncthreads();
    if (t == 0) esq[k] = wsum[0] + wsum[1] + wsum[2] + wsum[3];
}

// ---- codebook bf16 tile -> LDS (async if available, else sync copy) -------
__device__ __forceinline__ void load_cb_tile(const unsigned short* __restrict__ gsrc,
                                             unsigned short* __restrict__ ldst,
                                             int tid) {
    const int r0 = tid >> 6;            // 0..3
    const int c0 = (tid & 63) * 8;      // 0..504 step 8 (16B chunks)
    const unsigned short* g = gsrc + (size_t)r0 * D_DIM + c0;
    unsigned short* l = ldst + r0 * LDS_STRIDE + c0;
#if defined(HAVE_ASYNC_B128)
    #pragma unroll
    for (int i = 0; i < BN / 4; ++i) {
        __builtin_amdgcn_global_load_async_to_lds_b128(
            (v4i_as1*)(v4i*)(unsigned short*)g, (v4i_as3*)(v4i*)l, 0, 0);
        g += 4 * D_DIM;
        l += 4 * LDS_STRIDE;
    }
#else
    #pragma unroll
    for (int i = 0; i < BN / 4; ++i) {
        *(u32x4*)l = *(const u32x4*)g;
        g += 4 * D_DIM;
        l += 4 * LDS_STRIDE;
    }
#endif
}

// --------------------- fused distance GEMM + argmin -------------------------
extern "C" __global__ __launch_bounds__(256)
void vq_argmin_kernel(const float* __restrict__ z,
                      const unsigned short* __restrict__ cb_bf,
                      const float* __restrict__ esq,
                      int* __restrict__ idx_out, int K) {
    extern __shared__ unsigned char smem_raw[];
    unsigned short* zA  = (unsigned short*)smem_raw;      // BM x LDS_STRIDE bf16
    unsigned short* cB0 = zA + BM * LDS_STRIDE;           // double-buffered
    unsigned short* cB1 = cB0 + BN * LDS_STRIDE;          //   codebook tiles
    unsigned long long* amin = (unsigned long long*)(cB1 + BN * LDS_STRIDE);

    const int tid    = threadIdx.x;
    const int mblk   = blockIdx.x;
    const int lane   = tid & 31;
    const int wave   = tid >> 5;
    const int laneHi = lane >> 4;
    const int lmod   = lane & 15;

    if (tid < BM) amin[tid] = ~0ull;

    // ---- z block (BM x 512 fp32) -> bf16 LDS (converted once per element) --
    {
        const float* zbase = z + (size_t)mblk * BM * D_DIM;
        #pragma unroll
        for (int i = 0; i < (BM * D_DIM) / (256 * 4); ++i) {
            int li = (i * 256 + tid) * 4;
            int r  = li >> 9;
            int c  = li & 511;
            f32x4 v = *(const f32x4*)(zbase + r * D_DIM + c);
            u32x2 p;
            p.x = f2bf(v.x) | (f2bf(v.y) << 16);
            p.y = f2bf(v.z) | (f2bf(v.w) << 16);
            *(u32x2*)(zA + r * LDS_STRIDE + c) = p;
        }
    }

    // prologue: bring tile 0 into buffer 0
    load_cb_tile(cb_bf, cB0, tid);
    wait_async_all();
    __syncthreads();

    // wave tile assignment: 16-row strip, two 16-col tiles
    const int mt  = wave & 3;
    const int nt0 = (wave >> 2) * 2;
    const int rA  = mt  * 16 + lmod;
    const int rB0 = nt0 * 16 + lmod;
    const int rB1 = rB0 + 16;

    // ---- hoist all A fragments into registers (reused for every col tile) --
    union UA { u32x4 q[2]; v16bf v; };
    v16bf afrag[D_DIM / 32];
    #pragma unroll
    for (int d = 0; d < D_DIM / 32; ++d) {
        UA a;
        const unsigned short* ap = zA + rA * LDS_STRIDE + (d * 32 + laneHi * 8);
        a.q[0] = *(const u32x4*)ap;        // K = base + 0..7
        a.q[1] = *(const u32x4*)(ap + 16); // K = base + 16..23
        afrag[d] = a.v;
    }

    float minv[8];
    int   mini[8];
    #pragma unroll
    for (int r = 0; r < 8; ++r) { minv[r] = 3.4e38f; mini[r] = 0; }

    const int nKT = K / BN;
    for (int kt = 0; kt < nKT; ++kt) {
        unsigned short* cur = (kt & 1) ? cB1 : cB0;
        unsigned short* nxt = (kt & 1) ? cB0 : cB1;
        if (kt + 1 < nKT)   // async prefetch next tile while computing
            load_cb_tile(cb_bf + (size_t)(kt + 1) * BN * D_DIM, nxt, tid);

        const unsigned short* brow0 = cur + rB0 * LDS_STRIDE + laneHi * 8;
        const unsigned short* brow1 = cur + rB1 * LDS_STRIDE + laneHi * 8;

        v8f acc0 = {};
        v8f acc1 = {};
        // software-pipelined B fragments: load d+1 before issuing WMMAs for d
        UA b0c, b1c, b0n, b1n;
        b0c.q[0] = *(const u32x4*)brow0;
        b0c.q[1] = *(const u32x4*)(brow0 + 16);
        b1c.q[0] = *(const u32x4*)brow1;
        b1c.q[1] = *(const u32x4*)(brow1 + 16);
        #pragma unroll
        for (int d = 0; d < D_DIM / 32; ++d) {
            if (d + 1 < D_DIM / 32) {
                const unsigned short* bp0 = brow0 + (d + 1) * 32;
                const unsigned short* bp1 = brow1 + (d + 1) * 32;
                b0n.q[0] = *(const u32x4*)bp0;
                b0n.q[1] = *(const u32x4*)(bp0 + 16);
                b1n.q[0] = *(const u32x4*)bp1;
                b1n.q[1] = *(const u32x4*)(bp1 + 16);
            }
            acc0 = __builtin_amdgcn_wmma_f32_16x16x32_bf16(
                false, afrag[d], false, b0c.v, (short)0, acc0, false, false);
            acc1 = __builtin_amdgcn_wmma_f32_16x16x32_bf16(
                false, afrag[d], false, b1c.v, (short)0, acc1, false, false);
            b0c = b0n;
            b1c = b1n;
        }

        // dist = e_sq[col] - 2 * dot ; update running argmin
        const int col0 = kt * BN + nt0 * 16 + lmod;
        const int col1 = col0 + 16;
        const float e0 = esq[col0];
        const float e1 = esq[col1];
        #pragma unroll
        for (int r = 0; r < 8; ++r) {
            float d0 = __builtin_fmaf(-2.0f, acc0[r], e0);
            if (d0 < minv[r]) { minv[r] = d0; mini[r] = col0; }
            float d1 = __builtin_fmaf(-2.0f, acc1[r], e1);
            if (d1 < minv[r]) { minv[r] = d1; mini[r] = col1; }
        }

        wait_async_all();   // next tile resident
        __syncthreads();    // all waves done reading cur / issuing nxt
    }

    // cross-lane / cross-wave argmin via packed u64 LDS atomic min
    #pragma unroll
    for (int r = 0; r < 8; ++r) {
        int mloc = mt * 16 + r + laneHi * 8;   // C-layout row (VGPR r, lane half)
        unsigned long long key =
            ((unsigned long long)orderF(minv[r]) << 32) | (unsigned)mini[r];
        atomicMin(&amin[mloc], key);
    }
    __syncthreads();
    if (tid < BM) {
        idx_out[mblk * BM + tid] = (int)(amin[tid] & 0xffffffffu);
    }
}

// ------------------- gather + z_q + indices + commit loss -------------------
extern "C" __global__ void vq_init_loss_kernel(float* acc) { *acc = 0.0f; }

extern "C" __global__ __launch_bounds__(128)
void vq_gather_kernel(const float* __restrict__ z,
                      const float* __restrict__ cb,
                      const int* __restrict__ idx,
                      float* __restrict__ zq_out,
                      float* __restrict__ idx_out_f,
                      float* __restrict__ loss_acc) {
    const int m = blockIdx.x;
    const int t = threadIdx.x;
    const int code = idx[m];
    const float* zr = z  + (size_t)m    * D_DIM;
    const float* er = cb + (size_t)code * D_DIM;

    f32x4 zv = *(const f32x4*)(zr + t * 4);
    f32x4 ev = *(const f32x4*)(er + t * 4);
    *(f32x4*)(zq_out + (size_t)m * D_DIM + t * 4) = ev;   // z + sg(z_q - z) == z_q

    float dx = zv.x - ev.x, dy = zv.y - ev.y, dz = zv.z - ev.z, dw = zv.w - ev.w;
    float local = dx * dx + dy * dy + dz * dz + dw * dw;

    #pragma unroll
    for (int off = 16; off > 0; off >>= 1)
        local += __shfl_down(local, off, 32);
    __shared__ float wsum[4];
    if ((t & 31) == 0) wsum[t >> 5] = local;
    __syncthreads();
    if (t == 0) {
        atomicAdd(loss_acc, wsum[0] + wsum[1] + wsum[2] + wsum[3]);
        idx_out_f[m] = (float)code;
    }
}

extern "C" __global__ void vq_fin_loss_kernel(const float* acc, float* out,
                                              float inv_count) {
    *out = 0.25f * (*acc) * inv_count;   // COMMITMENT_WEIGHT * mean
}

// ---------------------------------------------------------------------------
extern "C" void kernel_launch(void* const* d_in, const int* in_sizes, int n_in,
                              void* d_out, int out_size, void* d_ws, size_t ws_size,
                              hipStream_t stream) {
    (void)n_in; (void)out_size; (void)ws_size;
    const float* z  = (const float*)d_in[0];   // [M, 512]
    const float* cb = (const float*)d_in[1];   // [K, 512]
    const int M = in_sizes[0] / D_DIM;         // 32768
    const int K = in_sizes[1] / D_DIM;         // 8192

    float* out_zq   = (float*)d_out;                  // M*512 floats
    float* out_idx  = out_zq + (size_t)M * D_DIM;     // M floats
    float* out_loss = out_idx + M;                    // 1 float

    unsigned short* cb_bf = (unsigned short*)d_ws;    // K*512 bf16 (8 MB)
    float* esq      = (float*)(cb_bf + (size_t)K * D_DIM);  // K floats
    int*   idx      = (int*)(esq + K);                // M ints
    float* loss_acc = (float*)(idx + M);              // 1 float

    vq_cvt_cb_kernel<<<K, 128, 0, stream>>>(cb, cb_bf, esq);

    size_t lds_bytes = (size_t)(BM + 2 * BN) * LDS_STRIDE * sizeof(unsigned short)
                     + (size_t)BM * sizeof(unsigned long long);
    vq_argmin_kernel<<<M / BM, 256, lds_bytes, stream>>>(z, cb_bf, esq, idx, K);

    vq_init_loss_kernel<<<1, 1, 0, stream>>>(loss_acc);
    vq_gather_kernel<<<M, 128, 0, stream>>>(z, cb, idx, out_zq, out_idx, loss_acc);
    vq_fin_loss_kernel<<<1, 1, 0, stream>>>(loss_acc, out_loss,
                                            1.0f / ((float)M * (float)D_DIM));
}
